// diffmodel_60584808677845
// MI455X (gfx1250) — matrix operand order repaired and checked by reading, not compiled, not run on hardware
//
#include <hip/hip_runtime.h>
#include <hip/hip_bf16.h>
#include <stdint.h>

// ---------------------------------------------------------------------------
// MI455X / gfx1250 implementation.
//  - All heavy math via v_wmma_f32_16x16x32_bf16 (f32 accumulate).
//  - scores refactored as x_l^T (Wq_h^T Wk_h) x_m  => one big GEMM instead of
//    two (saves ~155 GFLOP) and enables per-(batch,head) fusion in LDS.
//  - X tiles staged into LDS with GLOBAL_LOAD_ASYNC_TO_LDS_B128 (ASYNCcnt).
//  - Workspace use: ~40 MB.
// ---------------------------------------------------------------------------

#define USE_ASYNC_X 1   // flip to 0 to fall back to plain ds-store staging

typedef __attribute__((ext_vector_type(16))) __bf16        bfx16;
typedef __attribute__((ext_vector_type(8)))  float         fx8;
typedef __attribute__((ext_vector_type(4)))  unsigned int  u32x4;
typedef __attribute__((ext_vector_type(4)))  float         f32x4;

__device__ __forceinline__ unsigned short f2bf(float f) {
    unsigned int u = __float_as_uint(f);
    u += 0x7FFFu + ((u >> 16) & 1u);            // round-to-nearest-even
    return (unsigned short)(u >> 16);
}

union FragU { bfx16 v; u32x4 q[2]; };

__device__ __forceinline__ bfx16 ld_frag_pair(const unsigned short* p0,
                                              const unsigned short* p1) {
    FragU u;
    u.q[0] = *(const u32x4*)p0;
    u.q[1] = *(const u32x4*)p1;
    return u.v;
}

__device__ __forceinline__ fx8 fx8_zero() {
    fx8 v;
    for (int i = 0; i < 8; ++i) v[i] = 0.f;
    return v;
}

__device__ __forceinline__ fx8 wmma_bf16(bfx16 a, bfx16 b, fx8 c) {
    return __builtin_amdgcn_wmma_f32_16x16x32_bf16(
        /*neg_a=*/false, a, /*neg_b=*/false, b,
        /*c_mod=*/(short)0, c, /*reuse_a=*/false, /*reuse_b=*/false);
}

// ---------------------------------------------------------------------------
// Generic tiled GEMM: C(MxN) = op(A) * op(B) [+ bias], bf16 MACs, f32 accum.
//   A_TRANS=0: A stored (M,K) row-major.   A_TRANS=1: A stored (K,M) -> A^T*B.
//   B_TRANS=0: B stored (N,K) (C = A*B^T). B_TRANS=1: B stored (K,N) (C = A*B).
// blockDim = 256 (8 waves). Tile: 64(M) x 128(N) x 32(K).
// Requires M%64==0, N%128==0, K%32==0 (true for every call below).
// ---------------------------------------------------------------------------
template <bool A_TRANS, bool B_TRANS, bool OUT_BF16, bool ADD_BIAS>
__global__ __launch_bounds__(256)
void gemm_wmma_bf16(const float* __restrict__ A, int lda, long aZ,
                    const float* __restrict__ B, int ldb, long bZ,
                    void* __restrict__ Cv, int ldc, long cZ,
                    const float* __restrict__ bias,
                    int M, int N, int K)
{
    constexpr int BM = 64, BN = 128, BK = 32;
    __shared__ __align__(16) unsigned short Asd[BM][BK];
    __shared__ __align__(16) unsigned short Bsd[BN][BK];

    const int tid  = threadIdx.x;
    const int lane = tid & 31;
    const int wid  = tid >> 5;
    const int wr   = wid & 1;       // 2 row groups of 32 rows
    const int wc   = wid >> 1;      // 4 col groups of 32 cols
    const int z    = blockIdx.z;

    A += (long)z * aZ;
    B += (long)z * bZ;

    const int m0 = blockIdx.y * BM;
    const int n0 = blockIdx.x * BN;

    fx8 acc[2][2];
    for (int r = 0; r < 2; ++r)
        for (int c = 0; c < 2; ++c) acc[r][c] = fx8_zero();

    const int aoff = (lane >> 4) << 3;   // 0 or 8 halves
    const int boff = (lane >> 4) << 4;   // 0 or 16 halves
    const int cl   = lane & 15;

    for (int kc = 0; kc < K; kc += BK) {
        // ---- stage A tile -> Asd[m][k] (bf16) ----
        if constexpr (!A_TRANS) {
            const int row = tid >> 2;             // 0..63
            const int kq  = (tid & 3) * 8;        // 0,8,16,24
            const float* src = A + (long)(m0 + row) * lda + kc + kq;
            f32x4 v0 = *(const f32x4*)src;
            f32x4 v1 = *(const f32x4*)(src + 4);
            unsigned short* d = &Asd[row][kq];
            for (int i = 0; i < 4; ++i) { d[i] = f2bf(v0[i]); d[4 + i] = f2bf(v1[i]); }
        } else {
            const int kk = tid >> 3;              // 0..31
            const int mb = (tid & 7) * 8;
            const float* src = A + (long)(kc + kk) * lda + m0 + mb;
            for (int j = 0; j < 8; ++j) Asd[mb + j][kk] = f2bf(src[j]);
        }
        // ---- stage B tile -> Bsd[n][k] (bf16) ----
        if constexpr (!B_TRANS) {
            const int n  = tid >> 1;              // 0..127
            const int kq = (tid & 1) * 16;
            const float* src = B + (long)(n0 + n) * ldb + kc + kq;
            unsigned short* d = &Bsd[n][kq];
            for (int i = 0; i < 16; ++i) d[i] = f2bf(src[i]);
        } else {
            const int kk = tid >> 3;              // 0..31
            const int nb = (tid & 7) * 16;
            const float* src = B + (long)(kc + kk) * ldb + n0 + nb;
            for (int j = 0; j < 16; ++j) Bsd[nb + j][kk] = f2bf(src[j]);
        }
        __syncthreads();

        // ---- fragments + 4 WMMAs per wave ----
        bfx16 afr[2], bfr[2];
        for (int rt = 0; rt < 2; ++rt) {
            const unsigned short* pr = &Asd[wr * 32 + rt * 16 + cl][0];
            afr[rt] = ld_frag_pair(pr + aoff, pr + aoff + 16);
        }
        for (int ct = 0; ct < 2; ++ct) {
            const unsigned short* pc = &Bsd[wc * 32 + ct * 16 + cl][0];
            bfr[ct] = ld_frag_pair(pc + aoff, pc + aoff + 16);
        }
        for (int rt = 0; rt < 2; ++rt)
            for (int ct = 0; ct < 2; ++ct)
                acc[rt][ct] = wmma_bf16(afr[rt], bfr[ct], acc[rt][ct]);
        __syncthreads();
    }

    // ---- epilogue ----
    const int rbase = (lane >> 4) << 3;
    for (int rt = 0; rt < 2; ++rt) {
        for (int ct = 0; ct < 2; ++ct) {
            const int gr0 = m0 + wr * 32 + rt * 16 + rbase;
            const int gc  = n0 + wc * 32 + ct * 16 + cl;
            float bv = 0.f;
            if constexpr (ADD_BIAS) bv = bias[gc];
            for (int i = 0; i < 8; ++i) {
                const float v = acc[rt][ct][i] + bv;
                const long idx = (long)z * cZ + (long)(gr0 + i) * ldc + gc;
                if constexpr (OUT_BF16) ((unsigned short*)Cv)[idx] = f2bf(v);
                else                    ((float*)Cv)[idx]          = v;
            }
        }
    }
    (void)M; (void)N;
}

// ---------------------------------------------------------------------------
// Fused attention-integrate, one block per (batch b, head h):
//   P = Xb * M_h        (WMMA; Xb async-copied bf16 into LDS, M_h bf16 from L2)
//   S = scale * P*Xb^T  (WMMA, both operands from LDS)
//   row-softmax(S) summed over l -> atomicAdd into T[b][m] (summed over h).
// Mt layout: [head][f][e] bf16 where Mt[h][f][e] = sum_d Wq[h,d,e]*Wk[h,d,f],
// so B-matrix fragments (K=e contiguous) load straight off a row.
// Strides are in bf16 elements.
// ---------------------------------------------------------------------------
template <int HID, int LS>
__global__ __launch_bounds__(256)
void fused_integrate(const unsigned short* __restrict__ Xbf, long bStride, long rStride,
                     const unsigned short* __restrict__ Mt,
                     float* __restrict__ T)
{
    constexpr int ROW_T  = (LS + 15) / 16;
    constexpr int LS_PAD = ROW_T * 16;
    constexpr int COL_T  = HID / 16;
    constexpr int NW     = 8;
    constexpr int CT_W   = COL_T / NW;
    constexpr int KC     = HID / 32;

    __shared__ __align__(16) unsigned short Xs[LS_PAD][HID];
    __shared__ __align__(16) unsigned short Ps[LS_PAD][HID];
    __shared__ float Ss[LS_PAD][LS_PAD];
    __shared__ float accs[LS_PAD];

    const int tid  = threadIdx.x;
    const int lane = tid & 31;
    const int wid  = tid >> 5;
    const int b    = blockIdx.x;
    const int h    = blockIdx.y;

    if (tid < LS_PAD) accs[tid] = 0.f;
    // zero-pad rows >= LS (disjoint from async-copied region)
    for (int idx = LS * HID + tid; idx < LS_PAD * HID; idx += 256) Xs[0][idx] = 0;

#if USE_ASYNC_X
    {
        // Async byte-copy of Xb (LS x HID bf16) into LDS, 16B per lane-issue.
        constexpr int CHUNKS = LS * HID / 8;      // 16-byte chunks
        const unsigned long long sbase =
            (unsigned long long)(const void*)(Xbf + (long)b * bStride);
        const unsigned ldsbase = (unsigned)(size_t)(void*)&Xs[0][0];
        for (int c = tid; c < CHUNKS; c += 256) {
            const int l = (c * 8) / HID;
            const int d = (c * 8) % HID;
            const unsigned voff = (unsigned)(((long)l * rStride + d) * 2);
            const unsigned loff = ldsbase + (unsigned)((l * HID + d) * 2);
            asm volatile("global_load_async_to_lds_b128 %0, %1, %2"
                         :: "v"(loff), "v"(voff), "s"(sbase) : "memory");
        }
        asm volatile("s_wait_asynccnt 0x0" ::: "memory");
    }
#else
    for (int idx = tid; idx < LS * HID; idx += 256) {
        const int l = idx / HID;
        const int d = idx - l * HID;
        Xs[l][d] = Xbf[(long)b * bStride + (long)l * rStride + d];
    }
#endif
    __syncthreads();

    const unsigned short* Mh = Mt + (long)h * HID * HID;
    const int aoff  = (lane >> 4) << 3;
    const int boff  = (lane >> 4) << 4;
    const int cl    = lane & 15;
    const int rbase = (lane >> 4) << 3;

    // ---- P = Xb * M_h ----
    fx8 acc[ROW_T][CT_W];
    for (int rt = 0; rt < ROW_T; ++rt)
        for (int ct = 0; ct < CT_W; ++ct) acc[rt][ct] = fx8_zero();

    for (int kc = 0; kc < KC; ++kc) {
        const int k0 = kc * 32;
        bfx16 afr[ROW_T];
        for (int rt = 0; rt < ROW_T; ++rt) {
            const unsigned short* pr = &Xs[rt * 16 + cl][k0 + aoff];
            afr[rt] = ld_frag_pair(pr, pr + 16);
        }
        for (int ct = 0; ct < CT_W; ++ct) {
            const int c0 = (wid * CT_W + ct) * 16;
            const unsigned short* pb = Mh + (long)(c0 + cl) * HID + k0 + aoff;
            __builtin_prefetch(pb + 32, 0, 0);     // next k-chunk of M_h
            const bfx16 bfr = ld_frag_pair(pb, pb + 16);
            for (int rt = 0; rt < ROW_T; ++rt)
                acc[rt][ct] = wmma_bf16(afr[rt], bfr, acc[rt][ct]);
        }
    }
    for (int rt = 0; rt < ROW_T; ++rt)
        for (int ct = 0; ct < CT_W; ++ct) {
            const int c0 = (wid * CT_W + ct) * 16;
            for (int i = 0; i < 8; ++i)
                Ps[rt * 16 + rbase + i][c0 + cl] = f2bf(acc[rt][ct][i]);
        }
    __syncthreads();

    // ---- S = scale * P * Xb^T ----
    const float scale = rsqrtf((float)HID);
    if (wid < ROW_T * ROW_T) {
        const int rt = wid / ROW_T;
        const int mt = wid % ROW_T;
        fx8 s = fx8_zero();
        for (int kc = 0; kc < KC; ++kc) {
            const int k0 = kc * 32;
            const unsigned short* pa = &Ps[rt * 16 + cl][k0 + aoff];
            const bfx16 af = ld_frag_pair(pa, pa + 16);
            const unsigned short* pb = &Xs[mt * 16 + cl][k0 + aoff];
            const bfx16 bf = ld_frag_pair(pb, pb + 16);
            s = wmma_bf16(af, bf, s);
        }
        for (int i = 0; i < 8; ++i)
            Ss[rt * 16 + rbase + i][mt * 16 + cl] = s[i] * scale;
    }
    __syncthreads();

    // ---- per-row softmax over m, accumulate over l (this head) ----
    if (tid < LS) {
        const int l = tid;
        float mx = -3.4e38f;
        for (int m = 0; m < LS; ++m) mx = fmaxf(mx, Ss[l][m]);
        float se = 0.f;
        for (int m = 0; m < LS; ++m) se += __expf(Ss[l][m] - mx);
        const float inv = 1.f / se;
        for (int m = 0; m < LS; ++m)
            atomicAdd(&accs[m], __expf(Ss[l][m] - mx) * inv);
    }
    __syncthreads();
    if (tid < LS) atomicAdd(&T[(long)b * LS + tid], accs[tid]);
}

// ---------------------------------------------------------------------------
// Finalize: w = softmax(T/(RANK*LS)); pooled = sum_l w[l]*X[b,l,:]  (f32 X)
// ---------------------------------------------------------------------------
template <int HID, int LS>
__global__ __launch_bounds__(256)
void finalize_integrate(const float* __restrict__ T,
                        const float* __restrict__ X, long bStride, long rStride,
                        float* __restrict__ pooled, float* __restrict__ wOut)
{
    __shared__ float w[LS];
    const int b = blockIdx.x, tid = threadIdx.x;
    if (tid == 0) {
        float t[LS];
        float mx = -3.4e38f;
        for (int m = 0; m < LS; ++m) {
            t[m] = T[(long)b * LS + m] * (1.f / (16.f * (float)LS));
            mx = fmaxf(mx, t[m]);
        }
        float se = 0.f;
        for (int m = 0; m < LS; ++m) { t[m] = __expf(t[m] - mx); se += t[m]; }
        const float inv = 1.f / se;
        for (int m = 0; m < LS; ++m) w[m] = t[m] * inv;
    }
    __syncthreads();
    for (int d = tid; d < HID; d += 256) {
        float s = 0.f;
        for (int l = 0; l < LS; ++l)
            s += w[l] * X[(long)b * bStride + (long)l * rStride + d];
        pooled[(long)b * HID + d] = s;
    }
    if (wOut != nullptr && tid < LS) wOut[(long)b * LS + tid] = w[tid];
}

__global__ void zero_f32(float* __restrict__ p, long n) {
    const long i = (long)blockIdx.x * 256 + threadIdx.x;
    if (i < n) p[i] = 0.f;
}

__global__ void cvt_f32_bf16(const float* __restrict__ s,
                             unsigned short* __restrict__ d, long n) {
    const long i = (long)blockIdx.x * 256 + threadIdx.x;
    if (i < n) d[i] = f2bf(s[i]);
}

// ---------------------------------------------------------------------------
extern "C" void kernel_launch(void* const* d_in, const int* in_sizes, int n_in,
                              void* d_out, int out_size, void* d_ws, size_t ws_size,
                              hipStream_t stream)
{
    (void)in_sizes; (void)n_in; (void)out_size; (void)ws_size;

    const float* h_V    = (const float*)d_in[0];
    const float* h_EXV  = (const float*)d_in[1];
    const float* W_hV   = (const float*)d_in[2];
    const float* b_hV   = (const float*)d_in[3];
    const float* Wq_hv  = (const float*)d_in[4];
    const float* Wk_hv  = (const float*)d_in[5];
    const float* Wq_ce  = (const float*)d_in[6];
    const float* Wk_ce  = (const float*)d_in[7];
    const float* W_ce   = (const float*)d_in[8];
    const float* b_ce   = (const float*)d_in[9];
    const float* Wq_sce = (const float*)d_in[10];
    const float* Wk_sce = (const float*)d_in[11];

    constexpr int S = 4, L = 256, K = 32, OH = 128, EH = 256, XV = 384, R = 16;

    // workspace carve-up (~40 MB required)
    char* ws = (char*)d_ws;
    size_t off = 0;
    auto alloc = [&](size_t bytes) -> void* {
        void* p = ws + off;
        off = (off + bytes + 255) & ~(size_t)255;
        return p;
    };
    float*          hv     = (float*)alloc((size_t)S * L * EH * 4);
    float*          ce     = (float*)alloc((size_t)S * L * EH * 4);
    float*          cepool = (float*)alloc((size_t)S * L * XV * 4);
    unsigned short* MtHv   = (unsigned short*)alloc((size_t)R * EH * EH * 2);
    unsigned short* MtCe   = (unsigned short*)alloc((size_t)R * XV * XV * 2);
    unsigned short* MtSce  = (unsigned short*)alloc((size_t)R * EH * EH * 2);
    unsigned short* hvbf   = (unsigned short*)alloc((size_t)S * L * EH * 2);
    unsigned short* cebf   = (unsigned short*)alloc((size_t)S * L * EH * 2);
    unsigned short* hexvbf = (unsigned short*)alloc((size_t)S * L * K * XV * 2);
    const long nT = (long)L * S + (long)S * L * K + (long)L * S;
    float* Tall = (float*)alloc((size_t)nT * 4);
    float* Thv  = Tall;
    float* Tce  = Thv + L * S;
    float* Tsce = Tce + (long)S * L * K;

    const dim3 blk(256);
    auto cdiv = [](long n, long d) -> unsigned { return (unsigned)((n + d - 1) / d); };

    // 0) bf16 copy of h_EXV (for async LDS staging in the CE integrate)
    const long nEXV = (long)S * L * K * XV;
    cvt_f32_bf16<<<dim3(cdiv(nEXV, 256)), blk, 0, stream>>>(h_EXV, hexvbf, nEXV);

    // 1) hv = h_V @ W_hV^T + b_hV            (M=1024, N=256, K=128)
    gemm_wmma_bf16<false, false, false, true>
        <<<dim3(EH / 128, (S * L) / 64, 1), blk, 0, stream>>>(
            h_V, OH, 0, W_hV, OH, 0, (void*)hv, EH, 0, b_hV, S * L, EH, OH);
    cvt_f32_bf16<<<dim3(cdiv((long)S * L * EH, 256)), blk, 0, stream>>>(
        hv, hvbf, (long)S * L * EH);

    // 2) Gram matrices Mt[h][f][e] = (Wk_h^T Wq_h)  (bf16 out)
    gemm_wmma_bf16<true, true, true, false>
        <<<dim3(EH / 128, EH / 64, R), blk, 0, stream>>>(
            Wk_hv, EH, (long)EH * EH, Wq_hv, EH, (long)EH * EH,
            (void*)MtHv, EH, (long)EH * EH, nullptr, EH, EH, EH);
    gemm_wmma_bf16<true, true, true, false>
        <<<dim3(XV / 128, XV / 64, R), blk, 0, stream>>>(
            Wk_ce, XV, (long)XV * XV, Wq_ce, XV, (long)XV * XV,
            (void*)MtCe, XV, (long)XV * XV, nullptr, XV, XV, XV);
    gemm_wmma_bf16<true, true, true, false>
        <<<dim3(EH / 128, EH / 64, R), blk, 0, stream>>>(
            Wk_sce, EH, (long)EH * EH, Wq_sce, EH, (long)EH * EH,
            (void*)MtSce, EH, (long)EH * EH, nullptr, EH, EH, EH);

    // 3) zero the attention accumulators
    zero_f32<<<dim3(cdiv(nT, 256)), blk, 0, stream>>>(Tall, nT);

    float* out = (float*)d_out;
    float* out_hV_int = out;                       // (L, EH)
    float* out_hV_w   = out + (long)L * EH;        // (L, S)
    float* out_ce_int = out_hV_w + (long)L * S;    // (L, EH)
    float* out_ce_w   = out_ce_int + (long)L * EH; // (L, S)

    // 4) integrate hv over states (b = position; row s strided by L*EH)
    fused_integrate<EH, S><<<dim3(L, R), blk, 0, stream>>>(
        hvbf, (long)EH, (long)L * EH, MtHv, Thv);
    finalize_integrate<EH, S><<<dim3(L), blk, 0, stream>>>(
        Thv, hv, (long)EH, (long)L * EH, out_hV_int, out_hV_w);

    // 5) CE integrate over K neighbors (b = s*L+l; rows contiguous)
    fused_integrate<XV, K><<<dim3(S * L, R), blk, 0, stream>>>(
        hexvbf, (long)K * XV, (long)XV, MtCe, Tce);
    finalize_integrate<XV, K><<<dim3(S * L), blk, 0, stream>>>(
        Tce, h_EXV, (long)K * XV, (long)XV, cepool, nullptr);

    // 6) ce = cepool @ W_ce^T + b_ce          (M=1024, N=256, K=384)
    gemm_wmma_bf16<false, false, false, true>
        <<<dim3(EH / 128, (S * L) / 64, 1), blk, 0, stream>>>(
            cepool, XV, 0, W_ce, XV, 0, (void*)ce, EH, 0, b_ce, S * L, EH, XV);
    cvt_f32_bf16<<<dim3(cdiv((long)S * L * EH, 256)), blk, 0, stream>>>(
        ce, cebf, (long)S * L * EH);

    // 7) integrate ce over states
    fused_integrate<EH, S><<<dim3(L, R), blk, 0, stream>>>(
        cebf, (long)EH, (long)L * EH, MtSce, Tsce);
    finalize_integrate<EH, S><<<dim3(L), blk, 0, stream>>>(
        Tsce, ce, (long)EH, (long)L * EH, out_ce_int, out_ce_w);
}
